// SOPGraphSAGE_23673859736029
// MI455X (gfx1250) — matrix operand order repaired and checked by reading, not compiled, not run on hardware
//
#include <hip/hip_runtime.h>
#include <hip/hip_bf16.h>
#include <math.h>

#define EPSV 1e-5f

typedef unsigned int u32;
typedef unsigned short u16;
typedef __attribute__((ext_vector_type(4)))  u32    u32x4;
typedef __attribute__((ext_vector_type(16))) __bf16 v16bf;
typedef __attribute__((ext_vector_type(8)))  float  v8f;
typedef int i32x4 __attribute__((vector_size(16)));   // matches builtin's V4i

union Frag { u32x4 q[2]; v16bf v; };

__device__ __forceinline__ u16 f2bf(float f) {
    union { float f; u32 u; } v; v.f = f;
    u32 r = v.u + 0x7FFFu + ((v.u >> 16) & 1u);   // round-to-nearest-even
    return (u16)(r >> 16);
}
__device__ __forceinline__ float bf2f(u16 h) {
    union { u32 u; float f; } v; v.u = ((u32)h) << 16;
    return v.f;
}
__device__ __forceinline__ void atomicMaxF(float* addr, float val) {
    if (val >= 0.f) atomicMax((int*)addr, __float_as_int(val));
    else            atomicMin((u32*)addr, __float_as_uint(val));
}

// ---------------------------------------------------------------------------
// Stage one packed 128x128 bf16 weight (8192 dwords = 32KB) into LDS using the
// CDNA5 async global->LDS copy path (ASYNCcnt), falling back to a synchronous
// copy if the builtin is unavailable.  128 threads, 16 x b128 chunks each.
// ---------------------------------------------------------------------------
#if __has_builtin(__builtin_amdgcn_global_load_async_to_lds_b128)
#define HAS_ASYNC_LDS 1
#else
#define HAS_ASYNC_LDS 0
#endif

__device__ __forceinline__ void stage_b_to_lds(u32* bsh, const u32* Bp, int tid) {
#if HAS_ASYNC_LDS
#pragma unroll
    for (int i = 0; i < 16; i++) {
        int idx = (i * 128 + tid) * 4;     // dword index of this 16B chunk
        __builtin_amdgcn_global_load_async_to_lds_b128(
            (__attribute__((address_space(1))) i32x4*)(Bp + idx),
            (__attribute__((address_space(3))) i32x4*)(bsh + idx),
            0, 0);
    }
#if __has_builtin(__builtin_amdgcn_s_wait_asynccnt)
    __builtin_amdgcn_s_wait_asynccnt(0);
#else
    asm volatile("s_wait_asynccnt 0" ::: "memory");
#endif
#else
#pragma unroll
    for (int i = 0; i < 16; i++) {
        int idx = (i * 128 + tid) * 4;
        *(u32x4*)(bsh + idx) = *(const u32x4*)(Bp + idx);
    }
#endif
    __syncthreads();
}

// ---------------------------------------------------------------------------
// Pack a 128x128 row-major f32 weight into WMMA B-fragment order (bf16).
// Layout per (kt,nt) fragment tile: dword[lane*8 + j] = {K=kt*32+(lane>>4)*8+2j+(j>=4?8:0)
// (lo16), +1 (hi16)}, N = nt*16 + (lane&15).  grid = 32 blocks, block = 256.
// ---------------------------------------------------------------------------
__global__ __launch_bounds__(256) void pack_weight_bf16(const float* __restrict__ W,
                                                        u32* __restrict__ dst) {
    int ft   = blockIdx.x;            // kt*8 + nt
    int kt   = ft >> 3, nt = ft & 7;
    int lane = threadIdx.x >> 3;      // 0..31
    int j    = threadIdx.x & 7;       // 0..7
    int n    = nt * 16 + (lane & 15);
    int k    = kt * 32 + ((lane >> 4) * 8) + 2 * j + ((j >> 2) * 8);
    u16 lo = f2bf(W[(long)k * 128 + n]);
    u16 hi = f2bf(W[(long)(k + 1) * 128 + n]);
    dst[(long)ft * 256 + lane * 8 + j] = (u32)lo | ((u32)hi << 16);
}

// ---------------------------------------------------------------------------
// Node encoder: h = relu(LayerNorm(x @ ne_W + ne_b)).  block=128, grid=N.
// ---------------------------------------------------------------------------
__global__ __launch_bounds__(128) void node_encoder(
    const float* __restrict__ x, const float* __restrict__ W,
    const float* __restrict__ bias, const float* __restrict__ g,
    const float* __restrict__ beta, float* __restrict__ h,
    u16* __restrict__ hbf, int N_) {
    __shared__ float red[128];
    int n = blockIdx.x, c = threadIdx.x;
    float s = bias[c];
#pragma unroll
    for (int k = 0; k < 12; k++) s += x[(long)n * 12 + k] * W[k * 128 + c];
    red[c] = s; __syncthreads();
    for (int off = 64; off; off >>= 1) { if (c < off) red[c] += red[c + off]; __syncthreads(); }
    float mu = red[0] * (1.f / 128.f); __syncthreads();
    float d = s - mu;
    red[c] = d * d; __syncthreads();
    for (int off = 64; off; off >>= 1) { if (c < off) red[c] += red[c + off]; __syncthreads(); }
    float var = red[0] * (1.f / 128.f);
    float y = g[c] * d * rsqrtf(var + EPSV) + beta[c];
    y = fmaxf(y, 0.f);
    h[(long)n * 128 + c]   = y;
    hbf[(long)n * 128 + c] = f2bf(y);
}

// ---------------------------------------------------------------------------
// Edge MLP layer 1: t = relu(edge_attr @ W1 + b1), stored bf16.  grid=E.
// ---------------------------------------------------------------------------
__global__ __launch_bounds__(128) void edge_mlp1(
    const float* __restrict__ ea, const float* __restrict__ W1,
    const float* __restrict__ b1, u16* __restrict__ tbf, long E_) {
    long e = blockIdx.x; if (e >= E_) return;
    int c = threadIdx.x;
    float s = b1[c];
#pragma unroll
    for (int k = 0; k < 6; k++) s += ea[e * 6 + k] * W1[k * 128 + c];
    tbf[e * 128 + c] = f2bf(fmaxf(s, 0.f));
}

// ---------------------------------------------------------------------------
// Generic WMMA GEMM:  C[M x 128] = A_bf16[M x 128] @ Bpacked + bias.
// 4 waves/block, each wave owns a 16x128 row block (8 accumulator frags).
// B is staged into LDS once per workgroup via async copy; inner loop reads
// B fragments with ds_load_b128 and A fragments as two global b128 loads.
// ---------------------------------------------------------------------------
__global__ __launch_bounds__(128) void gemm128_wmma(
    const u16* __restrict__ A, const u32* __restrict__ Bp,
    const float* __restrict__ bias, float* __restrict__ Cf,
    u16* __restrict__ Cb, long M, int relu) {
    __shared__ u32 bsh[8192];          // 32KB packed weight
    stage_b_to_lds(bsh, Bp, threadIdx.x);

    int lane = threadIdx.x & 31, wave = threadIdx.x >> 5;
    long m0  = (long)blockIdx.x * 64 + wave * 16;
    int mrow = lane & 15;
    int kb   = (lane >> 4) * 8;

    v8f acc[8];
    v8f z8 = {0.f, 0.f, 0.f, 0.f, 0.f, 0.f, 0.f, 0.f};
#pragma unroll
    for (int i = 0; i < 8; i++) acc[i] = z8;

#pragma unroll
    for (int kt = 0; kt < 4; kt++) {
        Frag a;
        long row = m0 + mrow;
        if (row < M) {
            const u32x4* p0 = reinterpret_cast<const u32x4*>(A + row * 128 + kt * 32 + kb);
            a.q[0] = p0[0];   // K = kb .. kb+7
            a.q[1] = p0[2];   // K = kb+16 .. kb+23
        } else {
            u32x4 z4 = {0, 0, 0, 0}; a.q[0] = z4; a.q[1] = z4;
        }
        const u32* bbase = bsh + kt * 8 * 256 + lane * 8;
#pragma unroll
        for (int nt = 0; nt < 8; nt++) {
            Frag b;
            const u32x4* bp4 = reinterpret_cast<const u32x4*>(bbase + nt * 256);
            b.q[0] = bp4[0]; b.q[1] = bp4[1];
            acc[nt] = __builtin_amdgcn_wmma_f32_16x16x32_bf16(
                false, a.v, false, b.v, (short)0, acc[nt], false, false);
        }
    }

#pragma unroll
    for (int nt = 0; nt < 8; nt++) {
        int col = nt * 16 + (lane & 15);
        float bv = bias ? bias[col] : 0.f;
#pragma unroll
        for (int r = 0; r < 8; r++) {
            long row = m0 + (lane >> 4) * 8 + r;
            if (row < M) {
                float v = acc[nt][r] + bv;
                if (relu) v = fmaxf(v, 0.f);
                if (Cf) Cf[row * 128 + col] = v;
                if (Cb) Cb[row * 128 + col] = f2bf(v);
            }
        }
    }
}

// ---------------------------------------------------------------------------
// Self-loop attr: segment-mean of edge_emb over destination nodes (atomics).
// ---------------------------------------------------------------------------
__global__ __launch_bounds__(128) void loop_acc(
    const u16* __restrict__ eebf, const int* __restrict__ dstp,
    float* __restrict__ sum, float* __restrict__ deg, long E_) {
    long e = blockIdx.x; if (e >= E_) return;
    int c = threadIdx.x;
    int d = dstp[e];
    atomicAdd(&sum[(long)d * 128 + c], bf2f(eebf[e * 128 + c]));
    if (c == 0) atomicAdd(&deg[d], 1.f);
}

__global__ __launch_bounds__(256) void loop_finish(
    float* __restrict__ sum, const float* __restrict__ deg,
    u16* __restrict__ loopbf, int N_) {
    long i = (long)blockIdx.x * blockDim.x + threadIdx.x;
    if (i >= (long)N_ * 128) return;
    int n = (int)(i >> 7);
    loopbf[i] = f2bf(sum[i] / fmaxf(deg[n], 1.f));
    sum[i] = 0.f;   // re-zero: reused as the aggregation accumulator
}

// ---------------------------------------------------------------------------
// e_proj GEMM fused with GATv2 logits + segment max.
// Rows [0,E) come from edge_emb_bf16, rows [E,E+N) from loop_attr_bf16.
// Epilogue: m = A@We + xl[src] + xr[dst]; leaky(0.2); dot with att per head;
// write logits[row][head] and atomic-max into mx[dst][head].
// ---------------------------------------------------------------------------
__global__ __launch_bounds__(128) void gat_logits_wmma(
    const u16* __restrict__ Aedge, const u16* __restrict__ Aloop,
    const u32* __restrict__ Bp, const float* __restrict__ xl,
    const float* __restrict__ xr, const int* __restrict__ src,
    const int* __restrict__ dstp, const float* __restrict__ attv,
    float* __restrict__ logits, float* __restrict__ mx,
    long Etot, long Mtot) {
    __shared__ u32   bsh[8192];          // 32KB packed We
    __shared__ float smem[4][16][128];   // 32KB epilogue scratch
    stage_b_to_lds(bsh, Bp, threadIdx.x);

    int lane = threadIdx.x & 31, wave = threadIdx.x >> 5;
    long m0  = (long)blockIdx.x * 64 + wave * 16;
    int mrow = lane & 15;
    int kb   = (lane >> 4) * 8;

    v8f acc[8];
    v8f z8 = {0.f, 0.f, 0.f, 0.f, 0.f, 0.f, 0.f, 0.f};
#pragma unroll
    for (int i = 0; i < 8; i++) acc[i] = z8;

#pragma unroll
    for (int kt = 0; kt < 4; kt++) {
        Frag a;
        long row = m0 + mrow;
        const u16* ap = nullptr;
        if (row < Etot)      ap = Aedge + row * 128;
        else if (row < Mtot) ap = Aloop + (row - Etot) * 128;
        if (ap) {
            const u32x4* p0 = reinterpret_cast<const u32x4*>(ap + kt * 32 + kb);
            a.q[0] = p0[0]; a.q[1] = p0[2];
        } else {
            u32x4 z4 = {0, 0, 0, 0}; a.q[0] = z4; a.q[1] = z4;
        }
        const u32* bbase = bsh + kt * 8 * 256 + lane * 8;
#pragma unroll
        for (int nt = 0; nt < 8; nt++) {
            Frag b;
            const u32x4* bp4 = reinterpret_cast<const u32x4*>(bbase + nt * 256);
            b.q[0] = bp4[0]; b.q[1] = bp4[1];
            acc[nt] = __builtin_amdgcn_wmma_f32_16x16x32_bf16(
                false, a.v, false, b.v, (short)0, acc[nt], false, false);
        }
    }

    // Epilogue: gather xl[src]/xr[dst], leaky relu, scale by att, stash in LDS.
    int rbase = (lane >> 4) * 8;
    int sArr[8], dArr[8];
#pragma unroll
    for (int r = 0; r < 8; r++) {
        long g = m0 + rbase + r;
        if (g < Etot)      { sArr[r] = src[g]; dArr[r] = dstp[g]; }
        else if (g < Mtot) { int n = (int)(g - Etot); sArr[r] = n; dArr[r] = n; }
        else               { sArr[r] = 0; dArr[r] = 0; }
    }
#pragma unroll
    for (int nt = 0; nt < 8; nt++) {
        int col = nt * 16 + (lane & 15);
        float av = attv[col];
#pragma unroll
        for (int r = 0; r < 8; r++) {
            float m = acc[nt][r] + xl[(long)sArr[r] * 128 + col]
                                 + xr[(long)dArr[r] * 128 + col];
            m = (m > 0.f) ? m : 0.2f * m;
            smem[wave][rbase + r][col] = m * av;
        }
    }
    __syncthreads();

    // Reduce 32 channels per head: 16 rows x 4 heads = 64 sums per wave.
#pragma unroll
    for (int i = 0; i < 2; i++) {
        int idx = lane + 32 * i;      // 0..63
        int rr = idx >> 2, hh = idx & 3;
        float s = 0.f;
#pragma unroll
        for (int c = 0; c < 32; c++) s += smem[wave][rr][hh * 32 + c];
        long g = m0 + rr;
        if (g < Mtot) {
            logits[g * 4 + hh] = s;
            int dn = (g < Etot) ? dstp[g] : (int)(g - Etot);
            atomicMaxF(&mx[(long)dn * 4 + hh], s);
        }
    }
}

// ex = exp(logit - mx[dst]); den[dst] += ex  (in-place into logits buffer)
__global__ __launch_bounds__(256) void exden_kernel(
    float* __restrict__ logits, const float* __restrict__ mx,
    float* __restrict__ den, const int* __restrict__ dstp,
    long Etot, long Mtot) {
    long i = (long)blockIdx.x * blockDim.x + threadIdx.x;
    if (i >= Mtot * 4) return;
    long g = i >> 2; int hh = (int)(i & 3);
    int dn = (g < Etot) ? dstp[g] : (int)(g - Etot);
    float e = expf(logits[i] - mx[(long)dn * 4 + hh]);
    logits[i] = e;
    atomicAdd(&den[(long)dn * 4 + hh], e);
}

// out[dst] += (ex/den[dst]) * xl[src]
__global__ __launch_bounds__(128) void scatter_kernel(
    const float* __restrict__ ex, const float* __restrict__ den,
    const float* __restrict__ xl, const int* __restrict__ src,
    const int* __restrict__ dstp, float* __restrict__ out,
    long Etot, long Mtot) {
    long g = blockIdx.x; if (g >= Mtot) return;
    int c = threadIdx.x;
    int s, d;
    if (g < Etot) { s = src[g]; d = dstp[g]; } else { s = d = (int)(g - Etot); }
    int hh = c >> 5;
    float alpha = ex[g * 4 + hh] / den[(long)d * 4 + hh];
    atomicAdd(&out[(long)d * 128 + c], alpha * xl[(long)s * 128 + c]);
}

// Column reductions for GraphNorm / global mean (64 nodes per block).
__global__ __launch_bounds__(128) void colsum_kernel(
    const float* __restrict__ X, const float* __restrict__ cb,
    float* __restrict__ acc, int N_) {
    int c = threadIdx.x; int base = blockIdx.x * 64;
    float bv = cb ? cb[c] : 0.f;
    int lim = min(64, N_ - base);
    float s = 0.f;
    for (int i = 0; i < lim; i++) s += X[(long)(base + i) * 128 + c] + bv;
    atomicAdd(&acc[c], s);
}
__global__ __launch_bounds__(128) void colvar_kernel(
    const float* __restrict__ X, const float* __restrict__ cb,
    const float* __restrict__ mu, const float* __restrict__ a,
    float* __restrict__ acc, int N_) {
    int c = threadIdx.x; int base = blockIdx.x * 64;
    float bv = cb ? cb[c] : 0.f;
    float am = a[c] * mu[c];
    int lim = min(64, N_ - base);
    float s = 0.f;
    for (int i = 0; i < lim; i++) {
        float o = X[(long)(base + i) * 128 + c] + bv - am;
        s += o * o;
    }
    atomicAdd(&acc[c], s);
}
__global__ void scale_inplace(float* p, float s, int n) {
    int i = blockIdx.x * blockDim.x + threadIdx.x;
    if (i < n) p[i] *= s;
}
__global__ void rstd_kernel(float* v, float invN, int n) {
    int i = blockIdx.x * blockDim.x + threadIdx.x;
    if (i < n) v[i] = rsqrtf(v[i] * invN + EPSV);
}
// h += relu(GraphNorm(out + conv_b)); refresh h_bf16
__global__ __launch_bounds__(256) void gn_finalize(
    const float* __restrict__ X, const float* __restrict__ cb,
    const float* __restrict__ mu, const float* __restrict__ a,
    const float* __restrict__ rstd, const float* __restrict__ w,
    const float* __restrict__ b, float* __restrict__ h,
    u16* __restrict__ hbf, int N_) {
    long i = (long)blockIdx.x * blockDim.x + threadIdx.x;
    if (i >= (long)N_ * 128) return;
    int c = (int)(i & 127);
    float o = X[i] + cb[c] - a[c] * mu[c];
    float hn = fmaxf(w[c] * o * rstd[c] + b[c], 0.f);
    float nh = h[i] + hn;
    h[i]   = nh;
    hbf[i] = f2bf(nh);
}

__global__ void fill_f32(float* p, float v, long n) {
    long i = (long)blockIdx.x * blockDim.x + threadIdx.x;
    if (i < n) p[i] = v;
}
__global__ void copy_f32(const float* __restrict__ s, float* __restrict__ d, long n) {
    long i = (long)blockIdx.x * blockDim.x + threadIdx.x;
    if (i < n) d[i] = s[i];
}

// emb = LayerNorm(h @ emb_W + emb_b).  block=64, grid=N.
__global__ __launch_bounds__(64) void emb_head(
    const float* __restrict__ h, const float* __restrict__ W,
    const float* __restrict__ b, const float* __restrict__ g,
    const float* __restrict__ beta, float* __restrict__ out, int N_) {
    __shared__ float red[64];
    int n = blockIdx.x, t = threadIdx.x;
    const float* hr = h + (long)n * 128;
    float s = b[t];
    for (int d = 0; d < 128; d++) s += hr[d] * W[d * 64 + t];
    red[t] = s; __syncthreads();
    for (int off = 32; off; off >>= 1) { if (t < off) red[t] += red[t + off]; __syncthreads(); }
    float mu = red[0] * (1.f / 64.f); __syncthreads();
    float dd = s - mu;
    red[t] = dd * dd; __syncthreads();
    for (int off = 32; off; off >>= 1) { if (t < off) red[t] += red[t + off]; __syncthreads(); }
    float var = red[0] * (1.f / 64.f);
    out[(long)n * 64 + t] = g[t] * dd * rsqrtf(var + EPSV) + beta[t];
}

// 5 scalar heads: sigmoid(heads 0..3), relu(head 4).  block=64, grid=N.
__global__ __launch_bounds__(64) void scalar_heads(
    const float* __restrict__ h, const float* __restrict__ hW1,
    const float* __restrict__ hb1, const float* __restrict__ hW2,
    const float* __restrict__ hb2, float* __restrict__ d_out,
    int N_, long sc_off) {
    __shared__ float red[64];
    int n = blockIdx.x, t = threadIdx.x;
    const float* hr = h + (long)n * 128;
    for (int k = 0; k < 5; k++) {
        const float* W = hW1 + (long)k * 128 * 64;
        float z = hb1[k * 64 + t];
        for (int d = 0; d < 128; d++) z += hr[d] * W[d * 64 + t];
        z = fmaxf(z, 0.f);
        red[t] = z * hW2[k * 64 + t];
        __syncthreads();
        for (int off = 32; off; off >>= 1) { if (t < off) red[t] += red[t + off]; __syncthreads(); }
        if (t == 0) {
            float head = red[0] + hb2[k];
            float v = (k < 4) ? (1.f / (1.f + expf(-head))) : fmaxf(head, 0.f);
            d_out[sc_off + (long)k * N_ + n] = v;
        }
        __syncthreads();
    }
}

// global risk head: sigmoid(relu(mean(h) @ gW1 + gb1) @ gW2 + gb2)
__global__ __launch_bounds__(64) void net_head(
    const float* __restrict__ gsum, const float* __restrict__ gW1,
    const float* __restrict__ gb1, const float* __restrict__ gW2,
    const float* __restrict__ gb2, float* __restrict__ d_out,
    long off, float invN) {
    __shared__ float z[64];
    int t = threadIdx.x;
    float s = gb1[t];
    for (int d = 0; d < 128; d++) s += (gsum[d] * invN) * gW1[d * 64 + t];
    z[t] = fmaxf(s, 0.f);
    __syncthreads();
    if (t < 4) {
        float o = gb2[t];
        for (int k = 0; k < 64; k++) o += z[k] * gW2[k * 4 + t];
        d_out[off + t] = 1.f / (1.f + expf(-o));
    }
}

// ---------------------------------------------------------------------------
extern "C" void kernel_launch(void* const* d_in, const int* in_sizes, int n_in,
                              void* d_out_v, int out_size, void* d_ws, size_t ws_size,
                              hipStream_t stream) {
    const float* x         = (const float*)d_in[0];
    const float* edge_attr = (const float*)d_in[1];
    const int*   eidx      = (const int*)  d_in[2];
    const float* ne_W  = (const float*)d_in[3];
    const float* ne_b  = (const float*)d_in[4];
    const float* ne_g  = (const float*)d_in[5];
    const float* ne_be = (const float*)d_in[6];
    const float* ee_W1 = (const float*)d_in[7];
    const float* ee_b1 = (const float*)d_in[8];
    const float* ee_W2 = (const float*)d_in[9];
    const float* ee_b2 = (const float*)d_in[10];
    const float* Wl    = (const float*)d_in[11];
    const float* bl    = (const float*)d_in[12];
    const float* Wr    = (const float*)d_in[13];
    const float* br    = (const float*)d_in[14];
    const float* We    = (const float*)d_in[15];
    const float* att   = (const float*)d_in[16];
    const float* convb = (const float*)d_in[17];
    const float* gn_w  = (const float*)d_in[18];
    const float* gn_b  = (const float*)d_in[19];
    const float* gn_a  = (const float*)d_in[20];
    const float* embW  = (const float*)d_in[21];
    const float* embB  = (const float*)d_in[22];
    const float* embG  = (const float*)d_in[23];
    const float* embBe = (const float*)d_in[24];
    const float* hW1   = (const float*)d_in[25];
    const float* hb1   = (const float*)d_in[26];
    const float* hW2   = (const float*)d_in[27];
    const float* hb2   = (const float*)d_in[28];
    const float* gW1   = (const float*)d_in[29];
    const float* gb1   = (const float*)d_in[30];
    const float* gW2   = (const float*)d_in[31];
    const float* gb2   = (const float*)d_in[32];
    float* d_out = (float*)d_out_v;

    const long N = in_sizes[0] / 12;
    const long E = in_sizes[1] / 6;
    const long M = E + N;
    const int* src = eidx;
    const int* dst = eidx + E;

    // ---- workspace carve-out ----
    char* base = (char*)d_ws;
    size_t off = 0;
    auto carve = [&](size_t bytes) { size_t o = off; off = (off + bytes + 255) & ~(size_t)255; return o; };
    float* h       = (float*)(base + carve((size_t)N * 128 * 4));
    float* xl      = (float*)(base + carve((size_t)N * 128 * 4));
    float* xr      = (float*)(base + carve((size_t)N * 128 * 4));
    float* nodetmp = (float*)(base + carve((size_t)N * 128 * 4));   // loop-sum, then aggregation
    u16*   h_bf    = (u16*)  (base + carve((size_t)N * 128 * 2));
    u16*   loop_bf = (u16*)  (base + carve((size_t)N * 128 * 2));
    u16*   t_bf    = (u16*)  (base + carve((size_t)E * 128 * 2));
    u16*   ee_bf   = (u16*)  (base + carve((size_t)E * 128 * 2));
    float* logits  = (float*)(base + carve((size_t)M * 4 * 4));
    float* mx      = (float*)(base + carve((size_t)N * 4 * 4));
    float* den     = (float*)(base + carve((size_t)N * 4 * 4));
    float* deg     = (float*)(base + carve((size_t)N * 4));
    float* mu      = (float*)(base + carve(128 * 4));
    float* var     = (float*)(base + carve(128 * 4));
    float* gsum    = (float*)(base + carve(128 * 4));
    u32*   wpack   = (u32*)  (base + carve((size_t)10 * 8192 * 4));
    (void)ws_size; (void)n_in; (void)out_size;

    const long WP = 8192;  // dwords per packed 128x128 weight
    // ---- pack all 128x128 weights into WMMA B-fragment order ----
    pack_weight_bf16<<<32, 256, 0, stream>>>(ee_W2, wpack);
    for (int i = 0; i < 3; i++) {
        pack_weight_bf16<<<32, 256, 0, stream>>>(Wl + (long)i * 16384, wpack + (1 + i) * WP);
        pack_weight_bf16<<<32, 256, 0, stream>>>(Wr + (long)i * 16384, wpack + (4 + i) * WP);
        pack_weight_bf16<<<32, 256, 0, stream>>>(We + (long)i * 16384, wpack + (7 + i) * WP);
    }

    // ---- encoders ----
    node_encoder<<<(int)N, 128, 0, stream>>>(x, ne_W, ne_b, ne_g, ne_be, h, h_bf, (int)N);
    edge_mlp1<<<(int)E, 128, 0, stream>>>(edge_attr, ee_W1, ee_b1, t_bf, E);
    gemm128_wmma<<<(int)((E + 63) / 64), 128, 0, stream>>>(
        t_bf, wpack, ee_b2, nullptr, ee_bf, E, 0);

    const int gN = (int)((N + 63) / 64);
    const int gM = (int)((M + 63) / 64);

    // ---- 3 GATv2 + GraphNorm + residual layers ----
    for (int i = 0; i < 3; i++) {
        gemm128_wmma<<<gN, 128, 0, stream>>>(h_bf, wpack + (1 + i) * WP, bl + i * 128, xl, nullptr, N, 0);
        gemm128_wmma<<<gN, 128, 0, stream>>>(h_bf, wpack + (4 + i) * WP, br + i * 128, xr, nullptr, N, 0);

        fill_f32<<<(int)((N + 255) / 256), 256, 0, stream>>>(deg, 0.f, N);
        fill_f32<<<(int)((N * 128 + 255) / 256), 256, 0, stream>>>(nodetmp, 0.f, N * 128);
        loop_acc<<<(int)E, 128, 0, stream>>>(ee_bf, dst, nodetmp, deg, E);
        loop_finish<<<(int)((N * 128 + 255) / 256), 256, 0, stream>>>(nodetmp, deg, loop_bf, (int)N);

        fill_f32<<<(int)((N * 4 + 255) / 256), 256, 0, stream>>>(mx, -INFINITY, N * 4);
        fill_f32<<<(int)((N * 4 + 255) / 256), 256, 0, stream>>>(den, 0.f, N * 4);

        gat_logits_wmma<<<gM, 128, 0, stream>>>(
            ee_bf, loop_bf, wpack + (7 + i) * WP, xl, xr, src, dst,
            att + (long)i * 128, logits, mx, E, M);
        exden_kernel<<<(int)((M * 4 + 255) / 256), 256, 0, stream>>>(logits, mx, den, dst, E, M);
        scatter_kernel<<<(int)M, 128, 0, stream>>>(logits, den, xl, src, dst, nodetmp, E, M);

        fill_f32<<<1, 128, 0, stream>>>(mu, 0.f, 128);
        fill_f32<<<1, 128, 0, stream>>>(var, 0.f, 128);
        colsum_kernel<<<gN, 128, 0, stream>>>(nodetmp, convb + i * 128, mu, (int)N);
        scale_inplace<<<1, 128, 0, stream>>>(mu, 1.f / (float)N, 128);
        colvar_kernel<<<gN, 128, 0, stream>>>(nodetmp, convb + i * 128, mu, gn_a + i * 128, var, (int)N);
        rstd_kernel<<<1, 128, 0, stream>>>(var, 1.f / (float)N, 128);
        gn_finalize<<<(int)((N * 128 + 255) / 256), 256, 0, stream>>>(
            nodetmp, convb + i * 128, mu, gn_a + i * 128, var,
            gn_w + i * 128, gn_b + i * 128, h, h_bf, (int)N);
    }

    // ---- output heads ----
    const long sc_off  = N * 64;
    const long h_off   = N * 69;
    const long net_off = N * 197;
    emb_head<<<(int)N, 64, 0, stream>>>(h, embW, embB, embG, embBe, d_out, (int)N);
    scalar_heads<<<(int)N, 64, 0, stream>>>(h, hW1, hb1, hW2, hb2, d_out, (int)N, sc_off);
    copy_f32<<<(int)((N * 128 + 255) / 256), 256, 0, stream>>>(h, d_out + h_off, N * 128);
    fill_f32<<<1, 128, 0, stream>>>(gsum, 0.f, 128);
    colsum_kernel<<<gN, 128, 0, stream>>>(h, nullptr, gsum, (int)N);
    net_head<<<1, 64, 0, stream>>>(gsum, gW1, gb1, gW2, gb2, d_out, net_off, 1.f / (float)N);
}
